// MILSTMCell_70403103916353
// MI455X (gfx1250) — matrix operand order repaired and checked by reading, not compiled, not run on hardware
//
#include <hip/hip_runtime.h>
#include <hip/hip_bf16.h>

// ---------------------------------------------------------------------------
// MI-LSTM cell, fully fused for MI455X (gfx1250, wave32, WMMA).
//   gates = LN( gx + gh + gx*gh ),  gx = x@W + b,  gh = h@U
//   c' = LN( sig(f)*c + sig(i)*tanh(g) ),  h' = sig(o)*tanh(c')
//
// v3 changes vs v2:
//  - A staging buffer now stored in WMMA *fragment order*
//    (addr = kstep*1024 + slot*512 + lane*16): hot-loop A addresses are one
//    base VGPR + immediate ds offsets (no per-iteration XOR/shift VALU), and
//    accesses are effectively bank-conflict free.
//  - B double-buffer with two named register sets (no rotation copies ->
//    no v_dual_mov chains, no WMMA WAR v_nops).
//  - Bias folded into the MI-combine instead of accumulator broadcast.
// LDS budget: 256KB gate rows + 2x32KB A staging = 320KB (WGP max).
// ---------------------------------------------------------------------------

typedef __attribute__((ext_vector_type(16))) __bf16 v16bf;
typedef __attribute__((ext_vector_type(8)))  __bf16 v8bf;
typedef __attribute__((ext_vector_type(4)))  __bf16 v4bf;
typedef __attribute__((ext_vector_type(8)))  float  v8f;

#define BDIM 8192
#define KDIM 1024          // I == H == 1024
#define HDIM 1024
#define NDIM 4096          // 4H
#define OUTSEG (BDIM * HDIM)

#define G_BYTES    (16 * NDIM * 4)             // 262144: fp32 gate rows
#define AX_OFF     G_BYTES                     // 32KB x block, fragment order
#define AH_OFF     (G_BYTES + 16 * KDIM * 2)   // 32KB h block
#define SMEM_BYTES (G_BYTES + 2 * 16 * KDIM * 2)   // 327680 = 320KB

union ABFrag { v16bf v; v8bf h[2]; };

// ---- fp32 -> bf16 (vectorized x4) ----------------------------------------
__global__ void cvt_bf16_4(const float* __restrict__ in,
                           __bf16* __restrict__ out, int n4) {
  int i = blockIdx.x * blockDim.x + threadIdx.x;
  if (i < n4) {
    const float4 f = ((const float4*)in)[i];
    v4bf o = { (__bf16)f.x, (__bf16)f.y, (__bf16)f.z, (__bf16)f.w };
    ((v4bf*)out)[i] = o;
  }
}

// ---- fp32 [K,N] row-major -> bf16 [N,K] (transposed for B fragments) -----
__global__ void cvt_transpose_bf16(const float* __restrict__ in,
                                   __bf16* __restrict__ out, int K, int N) {
  int idx = blockIdx.x * blockDim.x + threadIdx.x;
  if (idx < K * N) {
    int n = idx / K, k = idx - n * K;
    out[idx] = (__bf16)in[k * N + n];
  }
}

__device__ __forceinline__ float sigmf(float x) {
  return 1.0f / (1.0f + __expf(-x));
}

__global__ __launch_bounds__(256, 1) void milstm_fused(
    const __bf16* __restrict__ xb, const __bf16* __restrict__ hb,
    const __bf16* __restrict__ Wt, const __bf16* __restrict__ Ut,
    const float* __restrict__ Wb,  const float* __restrict__ cin,
    const float* __restrict__ gng, const float* __restrict__ gnb,
    const float* __restrict__ cng, const float* __restrict__ cnb,
    float* __restrict__ out) {
  extern __shared__ char smem[];
  float* G = (float*)smem;                       // [16][NDIM]
  const int lane  = threadIdx.x & 31;
  const int wave  = threadIdx.x >> 5;            // 0..7, 512-col stripe each
  const int row0  = blockIdx.x * 16;
  const int lhalf = lane >> 4;                   // 0 | 1
  const int l15   = lane & 15;

  // ---- Phase 0: async-copy x/h row-blocks (16x1024 bf16 each) into LDS,
  // permuted into WMMA A-fragment order. Global 16B granule (row r, gk):
  //   kstep s = gk>>2, q = gk&3  ->  lane = r + 16*(q>>1), slot = q&1
  //   lds = s*1024 + slot*512 + lane*16
  {
    const char* xsrc = (const char*)(xb + (size_t)row0 * KDIM);
    const char* hsrc = (const char*)(hb + (size_t)row0 * KDIM);
    for (int j = 0; j < 8; ++j) {
      const int e  = (int)threadIdx.x + j * 256;  // granule id 0..2047
      const int r  = e >> 7;
      const int gk = e & 127;
      const int s  = gk >> 2, q = gk & 3;
      const uint32_t lds =
          (uint32_t)(s * 1024 + ((q & 1) << 9) + ((r + ((q >> 1) << 4)) << 4));
      const uint32_t lx = AX_OFF + lds;
      const uint32_t lh = AH_OFF + lds;
      const uint64_t gax = (uint64_t)(uintptr_t)(xsrc + (size_t)e * 16);
      const uint64_t gah = (uint64_t)(uintptr_t)(hsrc + (size_t)e * 16);
      asm volatile("global_load_async_to_lds_b128 %0, %1, off"
                   :: "v"(lx), "v"(gax) : "memory");
      asm volatile("global_load_async_to_lds_b128 %0, %1, off"
                   :: "v"(lh), "v"(gah) : "memory");
    }
    asm volatile("s_wait_asynccnt 0x0" ::: "memory");
  }
  __syncthreads();

  // ---- Phase 1: dual bf16 WMMA GEMM + multiplicative-integration combine --
  const int b0 = lhalf ? 16 : 0;                 // B frag K-half per lane
  const char* axbase = smem + AX_OFF + lane * 16; // A frag: base + imm offsets
  const char* ahbase = smem + AH_OFF + lane * 16;

  for (int t = 0; t < 32; ++t) {
    const int ncol = wave * 512 + t * 16 + l15;
    const __bf16* wrow = Wt + (size_t)ncol * KDIM;
    const __bf16* urow = Ut + (size_t)ncol * KDIM;

    v8f accx, acch;
    for (int i = 0; i < 8; ++i) { accx[i] = 0.0f; acch[i] = 0.0f; }

    // double-buffered B fragments (no rotation copies)
    ABFrag bw0, bu0, bw1, bu1;
    bw0.h[0] = *(const v8bf*)(wrow + b0);
    bw0.h[1] = *(const v8bf*)(wrow + b0 + 8);
    bu0.h[0] = *(const v8bf*)(urow + b0);
    bu0.h[1] = *(const v8bf*)(urow + b0 + 8);

#pragma clang loop unroll_count(2)
    for (int k = 0; k < KDIM; k += 64) {
      // prefetch B for k+32 into buffer 1
      bw1.h[0] = *(const v8bf*)(wrow + k + 32 + b0);
      bw1.h[1] = *(const v8bf*)(wrow + k + 32 + b0 + 8);
      bu1.h[0] = *(const v8bf*)(urow + k + 32 + b0);
      bu1.h[1] = *(const v8bf*)(urow + k + 32 + b0 + 8);
      // A fragments for k: immediate-offset LDS loads
      ABFrag ax0, ah0;
      ax0.h[0] = *(const v8bf*)(axbase + k * 32);
      ax0.h[1] = *(const v8bf*)(axbase + k * 32 + 512);
      ah0.h[0] = *(const v8bf*)(ahbase + k * 32);
      ah0.h[1] = *(const v8bf*)(ahbase + k * 32 + 512);
      accx = __builtin_amdgcn_wmma_f32_16x16x32_bf16(
          false, ax0.v, false, bw0.v, (short)0, accx, false, false);
      acch = __builtin_amdgcn_wmma_f32_16x16x32_bf16(
          false, ah0.v, false, bu0.v, (short)0, acch, false, false);
      // prefetch B for k+64 into buffer 0 (wrapped: branchless, in-bounds;
      // the wrapped value is dead past the last iteration)
      const int k2 = (k + 64) & (KDIM - 1);
      bw0.h[0] = *(const v8bf*)(wrow + k2 + b0);
      bw0.h[1] = *(const v8bf*)(wrow + k2 + b0 + 8);
      bu0.h[0] = *(const v8bf*)(urow + k2 + b0);
      bu0.h[1] = *(const v8bf*)(urow + k2 + b0 + 8);
      // A fragments for k+32
      ABFrag ax1, ah1;
      ax1.h[0] = *(const v8bf*)(axbase + k * 32 + 1024);
      ax1.h[1] = *(const v8bf*)(axbase + k * 32 + 1536);
      ah1.h[0] = *(const v8bf*)(ahbase + k * 32 + 1024);
      ah1.h[1] = *(const v8bf*)(ahbase + k * 32 + 1536);
      accx = __builtin_amdgcn_wmma_f32_16x16x32_bf16(
          false, ax1.v, false, bw1.v, (short)0, accx, false, false);
      acch = __builtin_amdgcn_wmma_f32_16x16x32_bf16(
          false, ah1.v, false, bu1.v, (short)0, acch, false, false);
    }

    // C/D fp32 layout: lane<16 -> M=i, lane>=16 -> M=8+i; N = lane&15
    const float bias = Wb[ncol];                 // gx = x@W + b
    const int mbase = lhalf * 8;
    for (int i = 0; i < 8; ++i) {
      const float gx = accx[i] + bias, gh = acch[i];
      G[(mbase + i) * NDIM + ncol] = gx + gh + gx * gh;   // MI combine
    }
  }
  __syncthreads();   // column-stripe writers -> row owners

  // ---- Phases 2-4: per-row LN + gates + cell update + LN + outputs --------
  for (int rr = 0; rr < 2; ++rr) {
    const int r = wave * 2 + rr;                 // each wave owns 2 rows
    float* grow = G + r * NDIM;

    float s1 = 0.f, s2 = 0.f;
    for (int j = lane; j < NDIM; j += 32) {
      const float v = grow[j]; s1 += v; s2 += v * v;
    }
    for (int m = 16; m; m >>= 1) {
      s1 += __shfl_xor(s1, m, 32); s2 += __shfl_xor(s2, m, 32);
    }
    const float mu   = s1 * (1.0f / NDIM);
    const float rstd = rsqrtf(s2 * (1.0f / NDIM) - mu * mu + 1e-5f);

    const float* crow = cin + (size_t)(row0 + r) * HDIM;
    float c1 = 0.f, c2 = 0.f;
    for (int j = lane; j < HDIM; j += 32) {
      const float iv = sigmf((grow[j]            - mu) * rstd * gng[j]            + gnb[j]);
      const float fv = sigmf((grow[HDIM + j]     - mu) * rstd * gng[HDIM + j]     + gnb[HDIM + j]);
      const float gv = tanhf((grow[2 * HDIM + j] - mu) * rstd * gng[2 * HDIM + j] + gnb[2 * HDIM + j]);
      const float ov = sigmf((grow[3 * HDIM + j] - mu) * rstd * gng[3 * HDIM + j] + gnb[3 * HDIM + j]);
      const float cn = fv * crow[j] + iv * gv;
      grow[j] = cn;                  // i-slot := c_new
      grow[HDIM + j] = ov;           // f-slot := o
      c1 += cn; c2 += cn * cn;
    }
    for (int m = 16; m; m >>= 1) {
      c1 += __shfl_xor(c1, m, 32); c2 += __shfl_xor(c2, m, 32);
    }
    const float cmu   = c1 * (1.0f / HDIM);
    const float crstd = rsqrtf(c2 * (1.0f / HDIM) - cmu * cmu + 1e-5f);

    const size_t obase = (size_t)(row0 + r) * HDIM;
    for (int j = lane; j < HDIM; j += 32) {
      const float cl = (grow[j] - cmu) * crstd * cng[j] + cnb[j];
      const float hn = grow[HDIM + j] * tanhf(cl);
      out[obase + j]              = hn;
      out[OUTSEG + obase + j]     = hn;
      out[2 * OUTSEG + obase + j] = cl;
    }
  }
}

extern "C" void kernel_launch(void* const* d_in, const int* in_sizes, int n_in,
                              void* d_out, int out_size, void* d_ws, size_t ws_size,
                              hipStream_t stream) {
  const float* x   = (const float*)d_in[0];
  const float* h   = (const float*)d_in[1];
  const float* c   = (const float*)d_in[2];
  const float* W_w = (const float*)d_in[3];
  const float* W_b = (const float*)d_in[4];
  const float* U_w = (const float*)d_in[5];
  const float* gng = (const float*)d_in[6];
  const float* gnb = (const float*)d_in[7];
  const float* cng = (const float*)d_in[8];
  const float* cnb = (const float*)d_in[9];
  float* out = (float*)d_out;

  // ws layout: xb(16MB) | hb(16MB) | Wt(8MB) | Ut(8MB) = 48MB bf16
  __bf16* xb = (__bf16*)d_ws;
  __bf16* hb = xb + (size_t)BDIM * KDIM;
  __bf16* Wt = hb + (size_t)BDIM * KDIM;
  __bf16* Ut = Wt + (size_t)NDIM * KDIM;

  cvt_bf16_4<<<(BDIM * KDIM / 4 + 255) / 256, 256, 0, stream>>>(x, xb, BDIM * KDIM / 4);
  cvt_bf16_4<<<(BDIM * KDIM / 4 + 255) / 256, 256, 0, stream>>>(h, hb, BDIM * KDIM / 4);
  cvt_transpose_bf16<<<(NDIM * KDIM + 255) / 256, 256, 0, stream>>>(W_w, Wt, KDIM, NDIM);
  cvt_transpose_bf16<<<(NDIM * KDIM + 255) / 256, 256, 0, stream>>>(U_w, Ut, KDIM, NDIM);

  milstm_fused<<<BDIM / 16, 256, SMEM_BYTES, stream>>>(
      xb, hb, Wt, Ut, W_b, c, gng, gnb, cng, cnb, out);
}
